// ModMultiScaleBlock_10677288698179
// MI455X (gfx1250) — matrix operand order repaired and checked by reading, not compiled, not run on hardware
//
#include <hip/hip_runtime.h>
#include <hip/hip_bf16.h>

typedef __bf16 bf16;
typedef __attribute__((ext_vector_type(16))) __bf16 v16bf;
typedef __attribute__((ext_vector_type(8)))  float  v8f;
typedef __attribute__((ext_vector_type(4)))  unsigned int u32x4;

#define DEVI __device__ __forceinline__

DEVI bf16 tobf(float f) {
  unsigned x = __float_as_uint(f);
  unsigned r = x + 0x7fffu + ((x >> 16) & 1u);   // round-to-nearest-even
  unsigned short hv = (unsigned short)(r >> 16);
  union { unsigned short u; bf16 b; } cv; cv.u = hv; return cv.b;
}

// CDNA5 async global->LDS DMA (ASYNCcnt-tracked, no VGPR staging).
DEVI void async_cp16(unsigned lds_off, const bf16* gaddr) {
  asm volatile("global_load_async_to_lds_b128 %0, %1, off"
               :: "v"(lds_off), "v"(gaddr) : "memory");
}
DEVI void wait_async0() {
  asm volatile("s_wait_asynccnt 0x0" ::: "memory");
}

// ---------------- WMMA fragment loaders (CDNA5 16-bit layouts) ----------------
// A (16x32, MxK): lane l holds row m=l&15; K chunks at g*8..g*8+7 and g*8+16..g*8+23 (g=l>>4)
// B (32x16, KxN): lane l holds col n=l&15; K chunks at g*8..g*8+7 and g*8+16..g*8+23 (g=l>>4)
struct Frag { union { v16bf v; u32x4 q[2]; }; };

DEVI void load_frag_a(Frag& f, const bf16* tile, int row0, int lane) {
  int r = row0 + (lane & 15);
  const bf16* p = tile + r * 32 + ((lane >> 4) << 3);
  f.q[0] = *(const u32x4*)p;
  f.q[1] = *(const u32x4*)(p + 16);
}
DEVI void load_frag_b(Frag& f, const bf16* tile, int row0, int lane) {
  int r = row0 + (lane & 15);
  const bf16* p = tile + r * 32 + ((lane >> 4) << 3);
  f.q[0] = *(const u32x4*)p;
  f.q[1] = *(const u32x4*)(p + 16);
}

// ---------------- Generic tiled WMMA GEMM, async double-buffered ----------------
// C[M,N] = A[M,K] * B[K,N], A row-major bf16, B n-major [N][K] bf16.
// BMODE: 0 = plain, 1 = attention QK^T offsets, 2 = attention P*V offsets.
// EPI:   0 = bf16 (+bias), 1 = f32 (+bias), 2 = GELU->bf16 (+bias),
//        3 = f32 * scale, 4 = bf16 (+bias) scattered to vT[win][dim][tok],
//        5 = f32 (+bias) + resid
template<int BM, int BN, int WM, int WN, int BMODE, int EPI>
__global__ __launch_bounds__(256) void gemm_k(
    const bf16* __restrict__ Ap, const bf16* __restrict__ Bp,
    const float* __restrict__ bias, const float* __restrict__ resid,
    void* __restrict__ Cp, int M, int N, int K,
    int lda, int ldb, int ldc, float scale)
{
  static_assert((BM / WM) * (BN / WN) == 8, "8 waves");
  __shared__ bf16 As[2][BM * 32];
  __shared__ bf16 Bs[2][BN * 32];

  const int tid  = threadIdx.x;
  const int lane = tid & 31;
  const int wv   = tid >> 5;
  constexpr int WC = BN / WN;
  const int waveM = (wv / WC) * WM;
  const int waveN = (wv % WC) * WN;
  const int mbase = blockIdx.y * BM;
  const int nbase = blockIdx.x * BN;

  long aoff = 0, boff = 0, coff = 0;
  if (BMODE == 1) {
    int z = blockIdx.z; int w = z / 12, hh = z % 12;
    aoff = (long)w * (256 * 768) + hh * 64; boff = aoff; coff = (long)z * 65536;
  }
  if (BMODE == 2) {
    int z = blockIdx.z; int w = z / 12, hh = z % 12;
    aoff = (long)z * 131072;                 // probs: bf16 rows packed in 1KB strides (lda=512)
    boff = ((long)w * 768 + hh * 64) * 256;
    coff = (long)w * (256 * 768) + hh * 64;
  }

  constexpr int MT = WM / 16, NT = WN / 16;
  v8f acc[MT][NT];
  #pragma unroll
  for (int i = 0; i < MT; i++)
    #pragma unroll
    for (int j = 0; j < NT; j++) {
      v8f z = {0.f, 0.f, 0.f, 0.f, 0.f, 0.f, 0.f, 0.f};
      acc[i][j] = z;
    }

  constexpr int ACH = (BM * 32) / (256 * 8);   // 16-byte chunks per thread
  constexpr int BCH = (BN * 32) / (256 * 8);

  auto issue_copies = [&](int k0, int buf) {
    #pragma unroll
    for (int i = 0; i < ACH; i++) {
      int c = tid + i * 256; int r = c >> 2; int kc = (c & 3) << 3;
      async_cp16((unsigned)(unsigned long long)&As[buf][r * 32 + kc],
                 Ap + aoff + (long)(mbase + r) * lda + (k0 + kc));
    }
    #pragma unroll
    for (int i = 0; i < BCH; i++) {
      int c = tid + i * 256; int r = c >> 2; int kc = (c & 3) << 3;
      async_cp16((unsigned)(unsigned long long)&Bs[buf][r * 32 + kc],
                 Bp + boff + (long)(nbase + r) * ldb + (k0 + kc));
    }
  };

  const int nk = K >> 5;
  issue_copies(0, 0);
  wait_async0();
  __syncthreads();

  for (int t = 0; t < nk; t++) {
    const int cur = t & 1;
    if (t + 1 < nk) issue_copies((t + 1) << 5, cur ^ 1);

    Frag af[MT], bfr[NT];
    #pragma unroll
    for (int i = 0; i < MT; i++) load_frag_a(af[i], As[cur], waveM + i * 16, lane);
    #pragma unroll
    for (int j = 0; j < NT; j++) load_frag_b(bfr[j], Bs[cur], waveN + j * 16, lane);
    #pragma unroll
    for (int i = 0; i < MT; i++)
      #pragma unroll
      for (int j = 0; j < NT; j++)
        acc[i][j] = __builtin_amdgcn_wmma_f32_16x16x32_bf16(
            false, af[i].v, false, bfr[j].v, (short)0, acc[i][j], false, false);

    wait_async0();     // next tile's async DMA landed
    __syncthreads();   // all waves done reading 'cur' (their wmma consumed the ds_loads)
  }

  // epilogue: C layout -> row m = e + 8*(lane>>4), col n = lane&15 (per 16x16 tile)
  bf16*  Cb = (bf16*)Cp;
  float* Cf = (float*)Cp;
  const int g8 = (lane >> 4) * 8;
  const int nl = lane & 15;
  #pragma unroll
  for (int i = 0; i < MT; i++) {
    #pragma unroll
    for (int j = 0; j < NT; j++) {
      int nc = nbase + waveN + j * 16 + nl;
      float bv = (bias != nullptr) ? bias[nc] : 0.f;
      int mt = mbase + waveM + i * 16 + g8;
      #pragma unroll
      for (int e = 0; e < 8; e++) {
        int mr = mt + e;
        float val = acc[i][j][e];
        if (EPI == 0) {
          Cb[coff + (long)mr * ldc + nc] = tobf(val + bv);
        } else if (EPI == 1) {
          Cf[coff + (long)mr * ldc + nc] = val + bv;
        } else if (EPI == 2) {
          float u = val + bv;
          float ge = 0.5f * u * (1.f + erff(u * 0.70710678118654752f));
          Cb[coff + (long)mr * ldc + nc] = tobf(ge);
        } else if (EPI == 3) {
          Cf[coff + (long)mr * ldc + nc] = val * scale;
        } else if (EPI == 4) {
          long vi = ((long)(mr >> 8) * 768 + nc) * 256 + (mr & 255);
          Cb[vi] = tobf(val + bv);
        } else if (EPI == 5) {
          Cf[coff + (long)mr * ldc + nc] = val + bv + resid[(long)mr * ldc + nc];
        }
      }
    }
  }
}

// ---------------- LayerNorm 1 + window-partition permute, fp32 -> bf16 ----------------
__global__ __launch_bounds__(256) void ln1_perm_k(
    const float* __restrict__ x, const float* __restrict__ g,
    const float* __restrict__ b, bf16* __restrict__ out)
{
  int row = blockIdx.x; int tid = threadIdx.x;
  const float* xr = x + (long)row * 768;
  float v0 = xr[tid], v1 = xr[tid + 256], v2 = xr[tid + 512];
  float s = v0 + v1 + v2;
  float s2 = v0 * v0 + v1 * v1 + v2 * v2;
  #pragma unroll
  for (int o = 16; o > 0; o >>= 1) { s += __shfl_xor(s, o, 32); s2 += __shfl_xor(s2, o, 32); }
  __shared__ float sh[16];
  if ((tid & 31) == 0) { sh[tid >> 5] = s; sh[8 + (tid >> 5)] = s2; }
  __syncthreads();
  s = 0.f; s2 = 0.f;
  #pragma unroll
  for (int w = 0; w < 8; w++) { s += sh[w]; s2 += sh[8 + w]; }
  float mu  = s * (1.f / 768.f);
  float var = s2 * (1.f / 768.f) - mu * mu;
  float rs  = rsqrtf(var + 1e-6f);
  int y = row >> 7, xc = row & 127;
  int orow = ((y >> 4) * 8 + (xc >> 4)) * 256 + (y & 15) * 16 + (xc & 15);
  bf16* o = out + (long)orow * 768;
  o[tid]       = tobf((v0 - mu) * rs * g[tid]       + b[tid]);
  o[tid + 256] = tobf((v1 - mu) * rs * g[tid + 256] + b[tid + 256]);
  o[tid + 512] = tobf((v2 - mu) * rs * g[tid + 512] + b[tid + 512]);
}

// ---------------- residual (un-permute) + LayerNorm 2 ----------------
__global__ __launch_bounds__(256) void ln2_k(
    const float* __restrict__ x, const float* __restrict__ proj,
    const float* __restrict__ g, const float* __restrict__ b,
    float* __restrict__ x2, bf16* __restrict__ h)
{
  int row = blockIdx.x; int tid = threadIdx.x;
  int y = row >> 7, xc = row & 127;
  int orow = ((y >> 4) * 8 + (xc >> 4)) * 256 + (y & 15) * 16 + (xc & 15);
  const float* xr = x + (long)row * 768;
  const float* pr = proj + (long)orow * 768;
  float v0 = xr[tid] + pr[tid];
  float v1 = xr[tid + 256] + pr[tid + 256];
  float v2 = xr[tid + 512] + pr[tid + 512];
  float* x2r = x2 + (long)row * 768;
  x2r[tid] = v0; x2r[tid + 256] = v1; x2r[tid + 512] = v2;
  float s = v0 + v1 + v2;
  float s2 = v0 * v0 + v1 * v1 + v2 * v2;
  #pragma unroll
  for (int o = 16; o > 0; o >>= 1) { s += __shfl_xor(s, o, 32); s2 += __shfl_xor(s2, o, 32); }
  __shared__ float sh[16];
  if ((tid & 31) == 0) { sh[tid >> 5] = s; sh[8 + (tid >> 5)] = s2; }
  __syncthreads();
  s = 0.f; s2 = 0.f;
  #pragma unroll
  for (int w = 0; w < 8; w++) { s += sh[w]; s2 += sh[8 + w]; }
  float mu  = s * (1.f / 768.f);
  float var = s2 * (1.f / 768.f) - mu * mu;
  float rs  = rsqrtf(var + 1e-6f);
  bf16* hr = h + (long)row * 768;
  hr[tid]       = tobf((v0 - mu) * rs * g[tid]       + b[tid]);
  hr[tid + 256] = tobf((v1 - mu) * rs * g[tid + 256] + b[tid + 256]);
  hr[tid + 512] = tobf((v2 - mu) * rs * g[tid + 512] + b[tid + 512]);
}

// -------- softmax: read 256 f32, write 256 bf16 in place over the row's first half --------
// (one wave per row; the wave reads its entire row before writing, rows are 1KB apart
//  so the bf16 output at the row base never overlaps another row's f32 data)
__global__ __launch_bounds__(256) void softmax_k(float* __restrict__ s)
{
  long row = (long)blockIdx.x * 8 + (threadIdx.x >> 5);
  int lane = threadIdx.x & 31;
  float* p = s + row * 256;
  float v[8]; float m = -3.4e38f;
  #pragma unroll
  for (int j = 0; j < 8; j++) { v[j] = p[lane + j * 32]; m = fmaxf(m, v[j]); }
  #pragma unroll
  for (int o = 16; o > 0; o >>= 1) m = fmaxf(m, __shfl_xor(m, o, 32));
  float sum = 0.f;
  #pragma unroll
  for (int j = 0; j < 8; j++) { v[j] = __expf(v[j] - m); sum += v[j]; }
  #pragma unroll
  for (int o = 16; o > 0; o >>= 1) sum += __shfl_xor(sum, o, 32);
  float inv = 1.f / sum;
  bf16* pb = (bf16*)p;
  #pragma unroll
  for (int j = 0; j < 8; j++) pb[lane + j * 32] = tobf(v[j] * inv);
}

// ---------------- weight convert + transpose: f32 [K][N] -> bf16 [N][K] ----------------
__global__ __launch_bounds__(256) void convt_k(
    const float* __restrict__ in, bf16* __restrict__ out, int K, int N)
{
  int i = blockIdx.x * 256 + threadIdx.x;
  int total = K * N;
  if (i < total) {
    int n = i / K; int k = i - n * K;
    out[i] = tobf(in[(long)k * N + n]);
  }
}

extern "C" void kernel_launch(void* const* d_in, const int* in_sizes, int n_in,
                              void* d_out, int out_size, void* d_ws, size_t ws_size,
                              hipStream_t stream)
{
  (void)in_sizes; (void)n_in; (void)out_size; (void)ws_size;
  const float* x   = (const float*)d_in[0];
  const float* Wq  = (const float*)d_in[1];
  const float* bq  = (const float*)d_in[2];
  const float* Wk  = (const float*)d_in[3];
  const float* bk  = (const float*)d_in[4];
  const float* Wv  = (const float*)d_in[5];
  const float* bv  = (const float*)d_in[6];
  const float* Wp  = (const float*)d_in[7];
  const float* bp  = (const float*)d_in[8];
  const float* g1  = (const float*)d_in[9];
  const float* bg1 = (const float*)d_in[10];
  const float* g2  = (const float*)d_in[11];
  const float* bg2 = (const float*)d_in[12];
  const float* W1  = (const float*)d_in[13];
  const float* bb1 = (const float*)d_in[14];
  const float* W2  = (const float*)d_in[15];
  const float* bb2 = (const float*)d_in[16];

  char* ws = (char*)d_ws;
  const size_t MBY = (size_t)1 << 20;
  bf16*  xnw  = (bf16*)(ws + 0);          // 24 MiB, window-ordered LN1 output
  bf16*  vT   = (bf16*)(ws + 24 * MBY);   // 24 MiB, [64][768][256]
  bf16*  q    = (bf16*)(ws + 48 * MBY);   // 24 MiB
  bf16*  kk   = (bf16*)(ws + 72 * MBY);   // 24 MiB
  bf16*  attn = (bf16*)(ws + 96 * MBY);   // 24 MiB
  float* sc   = (float*)(ws + 120 * MBY); // 192 MiB scores (fits MI455X 192MB L2)
  bf16*  h    = (bf16*)(ws + 312 * MBY);  // 24 MiB
  char*  wb   = ws + 336 * MBY;           // 13.5 MiB bf16 weights
  bf16* WqT = (bf16*)(wb);
  bf16* WkT = (bf16*)(wb + 1179648);
  bf16* WvT = (bf16*)(wb + 2359296);
  bf16* WpT = (bf16*)(wb + 3538944);
  bf16* W1T = (bf16*)(wb + 4718592);
  bf16* W2T = (bf16*)(wb + 9437184);
  // aliases (lifetimes disjoint):
  float* proj = (float*)(ws + 48 * MBY);  // over q,kk (dead after scores GEMM)
  float* x2   = (float*)(ws + 0);         // over xnw,vT (dead after P@V)
  bf16*  h1   = (bf16*)(ws + 120 * MBY);  // over scores (dead after P@V)
  float* out  = (float*)d_out;

  // weights -> bf16 [N][K]
  convt_k<<<dim3((768 * 768 + 255) / 256), 256, 0, stream>>>(Wq, WqT, 768, 768);
  convt_k<<<dim3((768 * 768 + 255) / 256), 256, 0, stream>>>(Wk, WkT, 768, 768);
  convt_k<<<dim3((768 * 768 + 255) / 256), 256, 0, stream>>>(Wv, WvT, 768, 768);
  convt_k<<<dim3((768 * 768 + 255) / 256), 256, 0, stream>>>(Wp, WpT, 768, 768);
  convt_k<<<dim3((768 * 3072 + 255) / 256), 256, 0, stream>>>(W1, W1T, 768, 3072);
  convt_k<<<dim3((768 * 3072 + 255) / 256), 256, 0, stream>>>(W2, W2T, 3072, 768);

  // LN1 + window partition
  ln1_perm_k<<<dim3(16384), 256, 0, stream>>>(x, g1, bg1, xnw);

  // Q, K, V projections (V stored transposed per window for P@V)
  gemm_k<128, 128, 32, 64, 0, 0><<<dim3(6, 128, 1), 256, 0, stream>>>(
      xnw, WqT, bq, nullptr, q, 16384, 768, 768, 768, 768, 768, 1.f);
  gemm_k<128, 128, 32, 64, 0, 0><<<dim3(6, 128, 1), 256, 0, stream>>>(
      xnw, WkT, bk, nullptr, kk, 16384, 768, 768, 768, 768, 768, 1.f);
  gemm_k<128, 128, 32, 64, 0, 4><<<dim3(6, 128, 1), 256, 0, stream>>>(
      xnw, WvT, bv, nullptr, vT, 16384, 768, 768, 768, 768, 768, 1.f);

  // scores = Q K^T * 1/sqrt(64), batched over 768 (win,head)
  gemm_k<128, 128, 32, 64, 1, 3><<<dim3(2, 2, 768), 256, 0, stream>>>(
      q, kk, nullptr, nullptr, sc, 256, 256, 64, 768, 768, 256, 0.125f);

  // softmax: f32 scores -> bf16 probs (in place, row stride 512 bf16)
  softmax_k<<<dim3(24576), 256, 0, stream>>>(sc);

  // out = P @ V  (A = bf16 probs at lda=512)
  gemm_k<128, 64, 32, 32, 2, 0><<<dim3(1, 2, 768), 256, 0, stream>>>(
      (const bf16*)sc, vT, nullptr, nullptr, attn, 256, 64, 256, 512, 256, 768, 1.f);

  // projection (f32 out, window order)
  gemm_k<128, 128, 32, 64, 0, 1><<<dim3(6, 128, 1), 256, 0, stream>>>(
      attn, WpT, bp, nullptr, proj, 16384, 768, 768, 768, 768, 768, 1.f);

  // x2 = x + unpermute(proj); h = LN2(x2)
  ln2_k<<<dim3(16384), 256, 0, stream>>>(x, proj, g2, bg2, x2, h);

  // MLP: GELU(h W1 + b1) W2 + b2 + x2
  gemm_k<128, 128, 32, 64, 0, 2><<<dim3(24, 128, 1), 256, 0, stream>>>(
      h, W1T, bb1, nullptr, h1, 16384, 3072, 768, 768, 768, 3072, 1.f);
  gemm_k<128, 128, 32, 64, 0, 5><<<dim3(6, 128, 1), 256, 0, stream>>>(
      h1, W2T, bb2, x2, out, 16384, 768, 3072, 3072, 3072, 768, 1.f);
}
